// ColBERTModel_502
// MI455X (gfx1250) — matrix operand order repaired and checked by reading, not compiled, not run on hardware
//
#include <hip/hip_runtime.h>
#include <hip/hip_bf16.h>
#include <math.h>

typedef __attribute__((ext_vector_type(16))) _Float16 v16h;
typedef __attribute__((ext_vector_type(8)))  _Float16 v8h;
typedef __attribute__((ext_vector_type(8)))  float    v8f;
typedef __attribute__((ext_vector_type(4)))  float    v4f;

#define HDIM 768
#define DDIM 128
#define ST   256
#define SL   32
#define BT   128
#define BL   128

// ---------------------------------------------------------------------------
// Kernel 0: W_proj [768,128] fp32  ->  Wt [128,768] f16 (transposed, so that
// WMMA B-fragments are contiguous 16-byte loads per lane).
// ---------------------------------------------------------------------------
__global__ void wt_convert_kernel(const float* __restrict__ W,
                                  _Float16* __restrict__ Wt) {
  int idx = blockIdx.x * blockDim.x + threadIdx.x;   // 0 .. 768*128-1
  int h = idx / DDIM;
  int d = idx % DDIM;
  Wt[(size_t)d * HDIM + h] = (_Float16)W[idx];
}

// ---------------------------------------------------------------------------
// Kernel 1: projection + bias + L2-normalize, f16 output.
// Block = 256 threads = 8 waves. Block computes 16 token rows x 128 cols.
// Wave w computes the 16x16 tile at columns w*16, accumulating K=768 in 24
// steps of v_wmma_f32_16x16x32_f16 (A converted fp32->f16 in-register).
// ---------------------------------------------------------------------------
__global__ __launch_bounds__(256) void proj_norm_kernel(
    const float* __restrict__ hidden,   // [ntok, 768] fp32
    const _Float16* __restrict__ Wt,    // [128, 768]  f16 (transposed W)
    const float* __restrict__ bias,     // [128] fp32
    _Float16* __restrict__ emb) {       // [ntok, 128] f16 (L2-normalized)
  __shared__ float tile[16][DDIM];      // 8 KB
  __shared__ float red[16][16];
  __shared__ float invn[16];

  const int tid  = threadIdx.x;
  const int wave = tid >> 5;
  const int lane = tid & 31;
  const int half = lane >> 4;
  const int lo   = lane & 15;           // A row-in-tile / B-C col-in-tile
  const int row0 = blockIdx.x * 16;
  const int n0   = wave * 16;

  const float*    arow = hidden + (size_t)(row0 + lo) * HDIM;
  const _Float16* bcol = Wt + (size_t)(n0 + lo) * HDIM;

  v8f acc = {};
  for (int k0 = 0; k0 < HDIM; k0 += 32) {
    // A fragment: rows = token row, K = k0..k0+31, fp32 -> f16
    v16h a;
    {
      const v4f* p = (const v4f*)(arow + k0 + 8 * half);
      v4f x0 = p[0], x1 = p[1];
      const v4f* q = (const v4f*)(arow + k0 + 16 + 8 * half);
      v4f y0 = q[0], y1 = q[1];
#pragma unroll
      for (int j = 0; j < 4; ++j) {
        a[j]      = (_Float16)x0[j];
        a[4 + j]  = (_Float16)x1[j];
        a[8 + j]  = (_Float16)y0[j];
        a[12 + j] = (_Float16)y1[j];
      }
    }
    // B fragment: column n0+lo of W (= row of Wt), contiguous K
    v16h b;
    {
      const v8h* p = (const v8h*)(bcol + k0 + 8 * half);
      const v8h* p2 = (const v8h*)(bcol + k0 + 16 + 8 * half);
      v8h b0 = p[0], b1 = p2[0];
#pragma unroll
      for (int j = 0; j < 8; ++j) { b[j] = b0[j]; b[8 + j] = b1[j]; }
    }
    acc = __builtin_amdgcn_wmma_f32_16x16x32_f16(false, a, false, b,
                                                 (short)0, acc, false, false);
  }

  // Scatter C tile (+bias) into LDS: lane holds col n0+lo, rows i+8*half.
  const float bv = bias[n0 + lo];
#pragma unroll
  for (int i = 0; i < 8; ++i)
    tile[i + 8 * half][n0 + lo] = acc[i] + bv;
  __syncthreads();

  // Row-wise L2 norm over 128 cols: 16 threads per row, 8 cols each.
  const int row = tid >> 4;
  const int seg = tid & 15;
  float s = 0.f;
#pragma unroll
  for (int j = 0; j < 8; ++j) {
    float v = tile[row][seg * 8 + j];
    s += v * v;
  }
  red[row][seg] = s;
  __syncthreads();
  if (tid < 16) {
    float t = 0.f;
#pragma unroll
    for (int j = 0; j < 16; ++j) t += red[tid][j];
    invn[tid] = 1.0f / fmaxf(sqrtf(t), 1e-12f);
  }
  __syncthreads();

  const float iv = invn[row];
  v8h o;
#pragma unroll
  for (int j = 0; j < 8; ++j)
    o[j] = (_Float16)(tile[row][seg * 8 + j] * iv);
  *(v8h*)(emb + (size_t)(row0 + row) * DDIM + seg * 8) = o;
}

// ---------------------------------------------------------------------------
// Kernel 2: zero the loss accumulator.
// ---------------------------------------------------------------------------
__global__ void init_kernel(float* out) { out[0] = 0.0f; }

// ---------------------------------------------------------------------------
// Kernel 3: maxsim + BCE. One block per (t,l) pair, 256 threads = 8 waves.
// Wave w owns text rows [w*32, w*32+32): 2 M-tiles x 2 N-tiles x 4 K-steps
// = 16 v_wmma_f32_16x16x32_f16 per wave. Then masked column-max, masked
// mean over label tokens, sigmoid output, BCE atomically accumulated.
// ---------------------------------------------------------------------------
__global__ __launch_bounds__(256) void maxsim_kernel(
    const _Float16* __restrict__ temb,    // [128*256, 128]
    const _Float16* __restrict__ lemb,    // [128*32, 128]
    const int* __restrict__ tmask,        // [128,256]
    const int* __restrict__ lmask,        // [128,32]
    const float* __restrict__ targets,    // [128,128]
    const float* __restrict__ scale_p,    // [1]
    float* __restrict__ out) {            // [1 + 128*128]
  __shared__ int   smt[ST];
  __shared__ int   sml[SL];
  __shared__ float part[16][SL];
  __shared__ float fin[SL];

  const int t    = blockIdx.x;
  const int l    = blockIdx.y;
  const int tid  = threadIdx.x;
  const int wave = tid >> 5;
  const int lane = tid & 31;
  const int half = lane >> 4;
  const int lo   = lane & 15;

  smt[tid] = tmask[t * ST + tid];
  if (tid < SL) sml[tid] = lmask[l * SL + tid];
  __syncthreads();

  const _Float16* A = temb + (size_t)t * ST * DDIM;   // [256,128] text tokens
  const _Float16* B = lemb + (size_t)l * SL * DDIM;   // [32,128]  label tokens

  const int mb = wave * 32;
  v8f acc00 = {}, acc01 = {}, acc10 = {}, acc11 = {};

#pragma unroll
  for (int k0 = 0; k0 < DDIM; k0 += 32) {
    // B fragments: N-tile 0 -> label tokens lo, N-tile 1 -> 16+lo.
    v16h b0, b1;
    {
      const v8h* p = (const v8h*)(B + (size_t)lo * DDIM + k0 + 8 * half);
      v8h x = p[0], y = p[2];
      const v8h* q = (const v8h*)(B + (size_t)(16 + lo) * DDIM + k0 + 8 * half);
      v8h u = q[0], v = q[2];
#pragma unroll
      for (int j = 0; j < 8; ++j) {
        b0[j] = x[j]; b0[8 + j] = y[j];
        b1[j] = u[j]; b1[8 + j] = v[j];
      }
    }
    // A fragments: M-tile 0 -> text rows mb+lo, M-tile 1 -> mb+16+lo.
    v16h a0, a1;
    {
      const _Float16* ar0 = A + (size_t)(mb + lo) * DDIM;
      const _Float16* ar1 = A + (size_t)(mb + 16 + lo) * DDIM;
      v8h x = *(const v8h*)(ar0 + k0 + 8 * half);
      v8h y = *(const v8h*)(ar0 + k0 + 16 + 8 * half);
      v8h u = *(const v8h*)(ar1 + k0 + 8 * half);
      v8h v = *(const v8h*)(ar1 + k0 + 16 + 8 * half);
#pragma unroll
      for (int j = 0; j < 8; ++j) {
        a0[j] = x[j]; a0[8 + j] = y[j];
        a1[j] = u[j]; a1[8 + j] = v[j];
      }
    }
    acc00 = __builtin_amdgcn_wmma_f32_16x16x32_f16(false, a0, false, b0, (short)0, acc00, false, false);
    acc01 = __builtin_amdgcn_wmma_f32_16x16x32_f16(false, a0, false, b1, (short)0, acc01, false, false);
    acc10 = __builtin_amdgcn_wmma_f32_16x16x32_f16(false, a1, false, b0, (short)0, acc10, false, false);
    acc11 = __builtin_amdgcn_wmma_f32_16x16x32_f16(false, a1, false, b1, (short)0, acc11, false, false);
  }

  // Masked max over this wave's 32 text rows.  C layout: lane holds column
  // lo (N-tile local), rows i + 8*half within each 16-row M-tile.
  const float NEG = -INFINITY;
  float mx0 = NEG, mx1 = NEG;
#pragma unroll
  for (int i = 0; i < 8; ++i) {
    const int s0 = mb + i + 8 * half;
    const int s1 = mb + 16 + i + 8 * half;
    float v00 = smt[s0] ? acc00[i] : NEG;
    float v10 = smt[s1] ? acc10[i] : NEG;
    mx0 = fmaxf(mx0, fmaxf(v00, v10));
    float v01 = smt[s0] ? acc01[i] : NEG;
    float v11 = smt[s1] ? acc11[i] : NEG;
    mx1 = fmaxf(mx1, fmaxf(v01, v11));
  }
  part[wave * 2 + half][lo]      = mx0;
  part[wave * 2 + half][16 + lo] = mx1;
  __syncthreads();

  if (tid < SL) {
    float mx = NEG;
#pragma unroll
    for (int p = 0; p < 16; ++p) mx = fmaxf(mx, part[p][tid]);
    fin[tid] = sml[tid] ? mx : 0.0f;
  }
  __syncthreads();

  if (tid == 0) {
    float sum = 0.f;
    int cnt = 0;
#pragma unroll
    for (int r = 0; r < SL; ++r) { sum += fin[r]; cnt += (sml[r] != 0); }
    if (cnt < 1) cnt = 1;
    const float logit = scale_p[0] * (sum / (float)cnt);
    const float sig = 1.0f / (1.0f + expf(-logit));
    out[1 + t * BL + l] = sig;
    const float tgt = targets[t * BL + l];
    const float bce = fmaxf(logit, 0.f) - logit * tgt + log1pf(expf(-fabsf(logit)));
    atomicAdd(out, bce * (1.0f / (float)(BT * BL)));
  }
}

// ---------------------------------------------------------------------------
extern "C" void kernel_launch(void* const* d_in, const int* in_sizes, int n_in,
                              void* d_out, int out_size, void* d_ws, size_t ws_size,
                              hipStream_t stream) {
  const float* text_hidden  = (const float*)d_in[0];  // [128,256,768]
  const float* label_hidden = (const float*)d_in[1];  // [128,32,768]
  const int*   text_mask    = (const int*)d_in[2];    // [128,256]
  const int*   label_mask   = (const int*)d_in[3];    // [128,32]
  const float* targets      = (const float*)d_in[4];  // [128,128]
  const float* W_proj       = (const float*)d_in[5];  // [768,128]
  const float* b_proj       = (const float*)d_in[6];  // [128]
  const float* score_scale  = (const float*)d_in[7];  // [1]
  float* out = (float*)d_out;                          // [1 + 128*128]

  char* ws = (char*)d_ws;
  _Float16* Wt   = (_Float16*)ws;                                   // 192 KB
  _Float16* temb = (_Float16*)(ws + 256 * 1024);                    // 8 MB
  _Float16* lemb = (_Float16*)(ws + 256 * 1024 + (size_t)BT * ST * DDIM * 2);

  wt_convert_kernel<<<(HDIM * DDIM) / 256, 256, 0, stream>>>(W_proj, Wt);

  proj_norm_kernel<<<(BT * ST) / 16, 256, 0, stream>>>(text_hidden, Wt, b_proj, temb);
  proj_norm_kernel<<<(BL * SL) / 16, 256, 0, stream>>>(label_hidden, Wt, b_proj, lemb);

  init_kernel<<<1, 1, 0, stream>>>(out);

  dim3 grid(BT, BL);
  maxsim_kernel<<<grid, 256, 0, stream>>>(temb, lemb, text_mask, label_mask,
                                          targets, score_scale, out);
}
